// ResidualCrossAttentionLayer_89446988906573
// MI455X (gfx1250) — compile-verified, matrix operand order checked by reading, not run on hardware
//
#include <hip/hip_runtime.h>

typedef __attribute__((ext_vector_type(16))) _Float16 v16h;
typedef __attribute__((ext_vector_type(8)))  float    v8f;
typedef unsigned int u32x4 __attribute__((ext_vector_type(4)));
typedef int          i32x8 __attribute__((ext_vector_type(8)));
typedef int          i32x4 __attribute__((ext_vector_type(4)));

// problem constants
static constexpr int BATCH = 32;
static constexpr int DIM   = 512;    // D
static constexpr int LEN   = 4096;   // L
static constexpr int NTOK  = 77;     // text tokens
static constexpr int D2    = 768;    // text dim
static constexpr int BC    = 24;     // conditioned batches
static constexpr int HEADS = 8;
static constexpr int DH    = 64;
static constexpr int MPAD  = 1856;   // BC*NTOK=1848 padded to 116*16
static constexpr int MROWS = BC * NTOK;

// ---- WMMA helpers -----------------------------------------------------------
__device__ __forceinline__ int frag_k(int j, int lane) {
  return (j < 8 ? j : j + 8) + ((lane & 16) ? 8 : 0);
}

__device__ __forceinline__ v16h load_frag_a(const _Float16* p, int lane) {
  p += (lane & 16) ? 8 : 0;
  v16h a;
#pragma unroll
  for (int j = 0; j < 8; ++j) { a[j] = p[j]; a[j + 8] = p[j + 16]; }
  return a;
}

__device__ __forceinline__ v16h load_frag_b(const _Float16* pack, int fragIdx, int lane) {
  const _Float16* p = pack + ((size_t)fragIdx * 32 + lane) * 16;
  v16h b;
#pragma unroll
  for (int j = 0; j < 16; ++j) b[j] = p[j];
  return b;
}

__device__ __forceinline__ v8f wmma16(v16h a, v16h b, v8f c) {
  return __builtin_amdgcn_wmma_f32_16x16x32_f16(false, a, false, b, (short)0, c, false, false);
}

// ---- TDM: async 2-D strided tile (global f32) -> linear LDS -----------------
// tile_d rows of tile_t contiguous f32, row stride row_stride (elements).
__device__ __forceinline__ void tdm_load_tile_f32(unsigned lds_off, const float* gsrc,
                                                  int tile_t, int tile_d, int row_stride) {
  unsigned long long ga = (unsigned long long)(size_t)gsrc;
  u32x4 g0;
  g0[0] = 1u;                                     // count=1, is_restore=0, gather off
  g0[1] = lds_off;                                // lds_addr
  g0[2] = (unsigned)(ga & 0xffffffffu);           // global_addr[31:0]
  g0[3] = (unsigned)((ga >> 32) & 0x1ffffffu)     // global_addr[56:32]
          | (2u << 30);                           // type=2 (image)
  i32x8 g1;
  g1[0] = (int)(2u << 16);                        // wg_mask=0, data_size=2 (4B)
  g1[1] = (int)(((unsigned)tile_t & 0xffffu) << 16);          // tensor_dim0 lo16 @ [63:48]
  g1[2] = (int)((((unsigned)tile_t >> 16) & 0xffffu)          // tensor_dim0 hi16
                | (((unsigned)tile_d & 0xffffu) << 16));      // tensor_dim1 lo16
  g1[3] = (int)((((unsigned)tile_d >> 16) & 0xffffu)          // tensor_dim1 hi16
                | (((unsigned)tile_t & 0xffffu) << 16));      // tile_dim0
  g1[4] = (int)((unsigned)tile_d & 0xffffu);                  // tile_dim1, tile_dim2=0
  g1[5] = (int)(unsigned)row_stride;                          // tensor_dim0_stride lo32
  g1[6] = (int)(((unsigned)row_stride >> 16) >> 16);          // stride0 hi16, stride1 lo16 = 0
  g1[7] = 0;                                                  // stride1 hi
  i32x4 gz4;
  gz4[0] = 0; gz4[1] = 0; gz4[2] = 0; gz4[3] = 0;
  i32x8 gz8;
#pragma unroll
  for (int i = 0; i < 8; ++i) gz8[i] = 0;
  __builtin_amdgcn_tensor_load_to_lds(g0, g1, gz4, gz4, gz8, 0);
}

// ---- pack a KxN f32 weight matrix into B-fragment f16 layout ----------------
__global__ void pack_b_kernel(const float* __restrict__ W, _Float16* __restrict__ out,
                              int Ncols, int NT, int total) {
  int idx = blockIdx.x * 256 + threadIdx.x;
  if (idx >= total) return;
  int j    = idx & 15;
  int lane = (idx >> 4) & 31;
  int rem  = idx >> 9;          // kk*NT + nt
  int nt   = rem % NT;
  int kk   = rem / NT;
  int k = kk * 32 + frag_k(j, lane);
  int n = nt * 16 + (lane & 15);
  out[idx] = (_Float16)W[(size_t)k * Ncols + n];
}

// ---- LayerNorm of gathered condition rows -> f16 ----------------------------
__global__ void cond_ln_kernel(const float* __restrict__ cond, const int* __restrict__ cidx,
                               const float* __restrict__ g, const float* __restrict__ bta,
                               _Float16* __restrict__ cn) {
  int row = blockIdx.x;                       // 0..MPAD-1
  int tid = threadIdx.x, lane = tid & 31, w = tid >> 5;
  _Float16* dst = cn + (size_t)row * D2;
  if (row >= MROWS) {                         // zero-fill pad rows
    for (int i = tid; i < D2; i += 256) dst[i] = (_Float16)0.f;
    return;
  }
  int b = row / NTOK, n = row % NTOK;
  const float* src = cond + ((size_t)cidx[b] * NTOK + n) * D2;
  __shared__ float red[16];
  float s = 0.f, ss = 0.f;
  for (int i = tid; i < D2; i += 256) { float x = src[i]; s += x; ss += x * x; }
  for (int o = 16; o >= 1; o >>= 1) { s += __shfl_xor(s, o, 32); ss += __shfl_xor(ss, o, 32); }
  if (lane == 0) { red[w] = s; red[8 + w] = ss; }
  __syncthreads();
  if (tid == 0) {
    float S = 0.f, SS = 0.f;
    for (int i = 0; i < 8; ++i) { S += red[i]; SS += red[8 + i]; }
    red[0] = S; red[8] = SS;
  }
  __syncthreads();
  float mu = red[0] * (1.f / D2);
  float var = red[8] * (1.f / D2) - mu * mu;
  float rs = rsqrtf(var + 1e-5f);
  for (int i = tid; i < D2; i += 256)
    dst[i] = (_Float16)((src[i] - mu) * rs * g[i] + bta[i]);
}

// ---- K/V projection via WMMA: [MPAD x 768] @ [768 x 512] + bias -------------
__global__ void __launch_bounds__(256) kv_kernel(const _Float16* __restrict__ cn,
                                                 const _Float16* __restrict__ wkpack,
                                                 const _Float16* __restrict__ wvpack,
                                                 const float* __restrict__ bk,
                                                 const float* __restrict__ bv,
                                                 float* __restrict__ kbuf,
                                                 float* __restrict__ vbuf) {
  int tid = threadIdx.x, lane = tid & 31, w = tid >> 5;
  int m0 = blockIdx.x * 16;
  int nt = blockIdx.y * 8 + w;               // 0..31
  bool isV = (blockIdx.z != 0);
  const _Float16* pack = isV ? wvpack : wkpack;
  const float* bias = isV ? bv : bk;
  float* obuf = isV ? vbuf : kbuf;
  int ln15 = lane & 15, hi8 = (lane >> 4) << 3;
  float bval = bias[nt * 16 + ln15];
  v8f acc;
#pragma unroll
  for (int i = 0; i < 8; ++i) acc[i] = bval;
  const _Float16* arow = cn + (size_t)(m0 + ln15) * D2;
  for (int kk = 0; kk < D2 / 32; ++kk) {
    v16h a = load_frag_a(arow + kk * 32, lane);
    v16h b = load_frag_b(pack, kk * 32 + nt, lane);
    acc = wmma16(a, b, acc);
  }
  // buffers padded to MPAD rows -> unconditional stores (no per-lane branches)
#pragma unroll
  for (int i = 0; i < 8; ++i) {
    int r = m0 + i + hi8;
    obuf[(size_t)r * DIM + nt * 16 + ln15] = acc[i];
  }
}

// ---- k softmax over tokens + context = k^T v, emitted in B-frag f16 ---------
__global__ void ctx_kernel(const float* __restrict__ kbuf, const float* __restrict__ vbuf,
                           _Float16* __restrict__ ctxpack) {
  int b = blockIdx.x, h = blockIdx.y, tid = threadIdx.x;
  __shared__ float kc[NTOK * 65];
  __shared__ float vc[NTOK * 65];
  for (int idx = tid; idx < NTOK * 64; idx += 256) {
    int n = idx >> 6, c = idx & 63;
    size_t g = ((size_t)(b * NTOK + n)) * DIM + h * 64 + c;
    kc[n * 65 + c] = kbuf[g];
    vc[n * 65 + c] = vbuf[g];
  }
  __syncthreads();
  if (tid < 64) {                             // softmax over n for column tid
    int c = tid;
    float m = -1e30f;
    for (int n = 0; n < NTOK; ++n) m = fmaxf(m, kc[n * 65 + c]);
    float s = 0.f;
    for (int n = 0; n < NTOK; ++n) { float e = __expf(kc[n * 65 + c] - m); kc[n * 65 + c] = e; s += e; }
    float inv = 1.f / s;
    for (int n = 0; n < NTOK; ++n) kc[n * 65 + c] *= inv;
  }
  __syncthreads();
  _Float16* dst = ctxpack + ((size_t)(b * HEADS + h)) * (DH * DH);
  for (int idx = tid; idx < DH * DH; idx += 256) {
    int j = idx & 15, lane = (idx >> 4) & 31, rem = idx >> 9; // rem = kk*4+nt
    int nt = rem & 3, kk = rem >> 2;
    int i = kk * 32 + frag_k(j, lane);        // context row (q-feature index)
    int jc = nt * 16 + (lane & 15);           // context col (v-feature index)
    float s = 0.f;
    for (int n = 0; n < NTOK; ++n) s += kc[n * 65 + i] * vc[n * 65 + jc];
    dst[idx] = (_Float16)s;
  }
}

// ---- fused LN -> Q GEMM -> softmax(dh) -> @context -> residual scatter ------
static constexpr int XS = 514;   // x tile stride (halves): conflict-free transpose writes
static constexpr int QS = 68;    // q staging stride (f32)
static constexpr int YS = 33;    // y staging stride (f32)
static constexpr int QH = 72;    // softmaxed-q f16 stride (mult of 8 -> aligned frags)
static constexpr int DCHUNK = 128;  // d-rows per TDM chunk

__global__ void __launch_bounds__(256) big_kernel(const float* __restrict__ in,
                                                  const int* __restrict__ cidx,
                                                  const float* __restrict__ lng,
                                                  const float* __restrict__ lnb,
                                                  const _Float16* __restrict__ wqpack,
                                                  const float* __restrict__ bq,
                                                  const _Float16* __restrict__ ctxpack,
                                                  float* __restrict__ out) {
  __shared__ float    stage[DCHUNK * 32];     // 16 KB TDM landing buffer
  __shared__ _Float16 xs[32 * XS];            // 32.9 KB
  __shared__ float    sbuf[32 * QS];          // 8.7 KB, aliased qs / ys
  __shared__ _Float16 qh[32 * QH];            // 4.6 KB
  int tid = threadIdx.x, lane = tid & 31, w = tid >> 5;
  int bi = blockIdx.y;
  int bsrc = cidx[bi];
  size_t t0 = (size_t)blockIdx.x * 32;
  const float* xin = in + (size_t)bsrc * DIM * LEN;
  float* xout = out + (size_t)bsrc * DIM * LEN;

  // P0: TDM-copy the strided [d][t] tile into LDS (4 chunks), convert to f16 [t][d]
  unsigned stage_off = (unsigned)(size_t)(&stage[0]);
  for (int c = 0; c < DIM / DCHUNK; ++c) {
    if (w == 0) {
      tdm_load_tile_f32(stage_off, xin + (size_t)(c * DCHUNK) * LEN + t0,
                        32, DCHUNK, LEN);
      __builtin_amdgcn_s_wait_tensorcnt(0);
    }
    __syncthreads();
#pragma unroll 4
    for (int it = 0; it < DCHUNK * 32 / 256; ++it) {
      int idx = it * 256 + tid;
      int d = idx >> 5, t = idx & 31;
      xs[t * XS + c * DCHUNK + d] = (_Float16)stage[d * 32 + t];
    }
    __syncthreads();
  }

  // P1: LayerNorm each of the 32 rows (one wave per row, 4 rounds)
  for (int r = 0; r < 4; ++r) {
    int t = w + r * 8;
    float s = 0.f, ss = 0.f;
#pragma unroll
    for (int i = 0; i < 16; ++i) { float x = (float)xs[t * XS + lane + 32 * i]; s += x; ss += x * x; }
    for (int o = 16; o >= 1; o >>= 1) { s += __shfl_xor(s, o, 32); ss += __shfl_xor(ss, o, 32); }
    float mu = s * (1.f / DIM);
    float var = ss * (1.f / DIM) - mu * mu;
    float rs = rsqrtf(var + 1e-5f);
#pragma unroll
    for (int i = 0; i < 16; ++i) {
      int d = lane + 32 * i;
      float x = (float)xs[t * XS + d];
      xs[t * XS + d] = (_Float16)((x - mu) * rs * lng[d] + lnb[d]);
    }
  }
  __syncthreads();

  int mi = w & 1, ni = w >> 1;                // wave -> (t-16-tile, n-16-tile)
  int ln15 = lane & 15, hi8 = (lane >> 4) << 3;
  int arow = mi * 16 + ln15;

  for (int h = 0; h < HEADS; ++h) {
    // GEMM1: q = LN(x) @ Wq + bq  (this head's 64 columns)
    float bval = bq[h * 64 + ni * 16 + ln15];
    v8f acc;
#pragma unroll
    for (int i = 0; i < 8; ++i) acc[i] = bval;
    for (int kk = 0; kk < DIM / 32; ++kk) {
      v16h a = load_frag_a(xs + arow * XS + kk * 32, lane);
      v16h b = load_frag_b(wqpack, kk * 32 + h * 4 + ni, lane);
      acc = wmma16(a, b, acc);
    }
#pragma unroll
    for (int i = 0; i < 8; ++i)
      sbuf[(mi * 16 + i + hi8) * QS + ni * 16 + ln15] = acc[i];
    __syncthreads();

    // softmax over the head's 64 columns (8 lanes per row)
    {
      int t = tid >> 3, gg = tid & 7;
      float vals[8];
      float m = -1e30f;
#pragma unroll
      for (int j = 0; j < 8; ++j) { vals[j] = sbuf[t * QS + gg * 8 + j]; m = fmaxf(m, vals[j]); }
      m = fmaxf(m, __shfl_xor(m, 1, 32));
      m = fmaxf(m, __shfl_xor(m, 2, 32));
      m = fmaxf(m, __shfl_xor(m, 4, 32));
      float sum = 0.f;
#pragma unroll
      for (int j = 0; j < 8; ++j) { vals[j] = __expf(vals[j] - m); sum += vals[j]; }
      sum += __shfl_xor(sum, 1, 32);
      sum += __shfl_xor(sum, 2, 32);
      sum += __shfl_xor(sum, 4, 32);
      float inv = 1.f / sum;
#pragma unroll
      for (int j = 0; j < 8; ++j) qh[t * QH + gg * 8 + j] = (_Float16)(vals[j] * inv);
    }
    __syncthreads();

    // GEMM2: y = q_sm @ context[h]   (K = 64)
    v8f acc2;
#pragma unroll
    for (int i = 0; i < 8; ++i) acc2[i] = 0.f;
    const _Float16* cpk = ctxpack + ((size_t)(bi * HEADS + h)) * (DH * DH);
    for (int kk = 0; kk < 2; ++kk) {
      v16h a = load_frag_a(qh + arow * QH + kk * 32, lane);
      v16h b = load_frag_b(cpk, kk * 4 + ni, lane);
      acc2 = wmma16(a, b, acc2);
    }
#pragma unroll
    for (int i = 0; i < 8; ++i)
      sbuf[(ni * 16 + ln15) * YS + mi * 16 + i + hi8] = acc2[i];
    __syncthreads();

    // residual scatter: out[b, h*64+n, t] = in[...] + y[n][t]  (coalesced over t)
    {
      int t = tid & 31, r0 = tid >> 5;
#pragma unroll
      for (int rr = 0; rr < 8; ++rr) {
        int nidx = rr * 8 + r0;
        size_t g = (size_t)(h * 64 + nidx) * LEN + t0 + t;
        xout[g] = xin[g] + sbuf[nidx * YS + t];
      }
    }
    __syncthreads();
  }
}

extern "C" void kernel_launch(void* const* d_in, const int* in_sizes, int n_in,
                              void* d_out, int out_size, void* d_ws, size_t ws_size,
                              hipStream_t stream) {
  (void)in_sizes; (void)n_in; (void)out_size; (void)ws_size;
  const float* input = (const float*)d_in[0];
  const float* cond  = (const float*)d_in[1];
  const int*   cidx  = (const int*)d_in[2];
  const float* ln_g  = (const float*)d_in[3];
  const float* ln_b  = (const float*)d_in[4];
  const float* tln_g = (const float*)d_in[5];
  const float* tln_b = (const float*)d_in[6];
  const float* Wq    = (const float*)d_in[7];
  const float* bq    = (const float*)d_in[8];
  const float* Wk    = (const float*)d_in[9];
  const float* bk    = (const float*)d_in[10];
  const float* Wv    = (const float*)d_in[11];
  const float* bv    = (const float*)d_in[12];
  float* out = (float*)d_out;

  char* ws = (char*)d_ws;
  auto alloc = [&](size_t bytes) {
    char* p = ws;
    ws += (bytes + 255) & ~(size_t)255;
    return p;
  };
  _Float16* wqpack  = (_Float16*)alloc((size_t)DIM * DIM * 2);
  _Float16* wkpack  = (_Float16*)alloc((size_t)D2 * DIM * 2);
  _Float16* wvpack  = (_Float16*)alloc((size_t)D2 * DIM * 2);
  _Float16* cn      = (_Float16*)alloc((size_t)MPAD * D2 * 2);
  float*    kbuf    = (float*)alloc((size_t)MPAD * DIM * 4);   // padded rows
  float*    vbuf    = (float*)alloc((size_t)MPAD * DIM * 4);   // padded rows
  _Float16* ctxpack = (_Float16*)alloc((size_t)BC * HEADS * DH * DH * 2);

  // base copy input -> out (conditioned region overwritten by big_kernel later)
  (void)hipMemcpyAsync(out, input, (size_t)BATCH * DIM * LEN * sizeof(float),
                       hipMemcpyDeviceToDevice, stream);

  int totq = DIM * DIM, totk = D2 * DIM;
  pack_b_kernel<<<(totq + 255) / 256, 256, 0, stream>>>(Wq, wqpack, DIM, DIM / 16, totq);
  pack_b_kernel<<<(totk + 255) / 256, 256, 0, stream>>>(Wk, wkpack, DIM, DIM / 16, totk);
  pack_b_kernel<<<(totk + 255) / 256, 256, 0, stream>>>(Wv, wvpack, DIM, DIM / 16, totk);

  cond_ln_kernel<<<MPAD, 256, 0, stream>>>(cond, cidx, tln_g, tln_b, cn);
  kv_kernel<<<dim3(MPAD / 16, 4, 2), 256, 0, stream>>>(cn, wkpack, wvpack, bk, bv, kbuf, vbuf);
  ctx_kernel<<<dim3(BC, HEADS), 256, 0, stream>>>(kbuf, vbuf, ctxpack);

  big_kernel<<<dim3(LEN / 32, BC), 256, 0, stream>>>(input, cidx, ln_g, ln_b,
                                                     wqpack, bq, ctxpack, out);
}
